// Equivariant_layer_21638045237877
// MI455X (gfx1250) — compile-verified, matrix operand order checked by reading, not run on hardware
//
#include <hip/hip_runtime.h>
#include <hip/hip_bf16.h>
#include <math.h>

// ---------------------------------------------------------------------------
// CDNA5 (gfx1250) WMMA types & fragment helpers
// ---------------------------------------------------------------------------
typedef __attribute__((ext_vector_type(16))) __bf16 v16bf;
typedef __attribute__((ext_vector_type(8)))  float  v8f;

__device__ __forceinline__ v8f wmma_bf16(v16bf a, v16bf b, v8f c) {
  // (neg_a, A, neg_b, B, c_mod, C, reuse_a, reuse_b) -> v_wmma_f32_16x16x32_bf16
  return __builtin_amdgcn_wmma_f32_16x16x32_bf16(false, a, false, b, (short)0, c,
                                                 false, false);
}

// 16-bit A fragment 16x32 (MxK). Lane = 16*half + m holds row m.
// Packing = two contiguous 8-element runs: k in [k0+8h, +8) and [k0+16+8h, +8).
__device__ __forceinline__ v16bf
load_a_frag(const float* __restrict__ base, int row_stride, int elem_stride,
            int row0, int nrows, int k0, int Keff, const float* __restrict__ zscale) {
  int l = threadIdx.x & 31;
  int m = l & 15, half = l >> 4;
  int row = row0 + m;
  v16bf a;
  if (row < nrows) {
    float zs = zscale ? zscale[row] : 1.f;
    const float* rp = base + (size_t)row * row_stride;
#pragma unroll
    for (int i = 0; i < 8; ++i) {
      int ka = k0 + 8 * half + i;
      int kb = k0 + 16 + 8 * half + i;
      a[i]     = (ka < Keff) ? (__bf16)(rp[(size_t)ka * elem_stride] * zs) : (__bf16)0.f;
      a[8 + i] = (kb < Keff) ? (__bf16)(rp[(size_t)kb * elem_stride] * zs) : (__bf16)0.f;
    }
  } else {
#pragma unroll
    for (int i = 0; i < 16; ++i) a[i] = (__bf16)0.f;
  }
  return a;
}

// Generic (slow) B fragment 32x16 (KxN) from row-major f32 — used only by the
// one-shot weight pre-swizzle kernel.
__device__ __forceinline__ v16bf
load_b_frag(const float* __restrict__ B, int ldb, int k0, int Keff,
            int col0, int ncols) {
  int l = threadIdx.x & 31;
  int n = l & 15, half = l >> 4;
  int col = col0 + n;
  bool cv = col < ncols;
  v16bf b;
#pragma unroll
  for (int i = 0; i < 8; ++i) {
    int ka = k0 + 8 * half + i;
    int kb = k0 + 16 + 8 * half + i;
    b[i]     = (cv && ka < Keff) ? (__bf16)B[(size_t)ka * ldb + col] : (__bf16)0.f;
    b[8 + i] = (cv && kb < Keff) ? (__bf16)B[(size_t)kb * ldb + col] : (__bf16)0.f;
  }
  return b;
}

// Fast B fragment: one aligned 32-byte v16bf load from the pre-swizzled buffer
// (layout: [block][lane][16 bf16]).
__device__ __forceinline__ v16bf load_b_swz(const __bf16* __restrict__ base, int blk) {
  return *(const v16bf*)(base + ((size_t)blk * 32 + (threadIdx.x & 31)) * 16);
}

// f32 C/D store: VGPR r -> row r + 8*half, lane -> col (l&15).
// Wave-uniform full-tile fast path avoids per-element exec branching.
__device__ __forceinline__ void
store_c(float* __restrict__ dst, int row_stride, int elem_stride,
        int row0, int nrows, int col0, v8f acc, float scale,
        const float* __restrict__ addp, float addscale) {
  int l = threadIdx.x & 31;
  int n = l & 15, half = l >> 4;
  int col = col0 + n;
  if (row0 + 16 <= nrows) {
#pragma unroll
    for (int r = 0; r < 8; ++r) {
      size_t idx = (size_t)(row0 + r + 8 * half) * row_stride + (size_t)col * elem_stride;
      float v = acc[r] * scale;
      if (addp) v += addscale * addp[idx];
      dst[idx] = v;
    }
  } else {
#pragma unroll
    for (int r = 0; r < 8; ++r) {
      int row = row0 + r + 8 * half;
      if (row < nrows) {
        size_t idx = (size_t)row * row_stride + (size_t)col * elem_stride;
        float v = acc[r] * scale;
        if (addp) v += addscale * addp[idx];
        dst[idx] = v;
      }
    }
  }
}

__device__ __forceinline__ float silu_f(float x) { return x / (1.f + __expf(-x)); }

// ---------------------------------------------------------------------------
// K0: zero the scatter accumulators
// ---------------------------------------------------------------------------
__global__ void zero_kernel(float* __restrict__ p, int n) {
  int i = blockIdx.x * blockDim.x + threadIdx.x;
  if (i < n) p[i] = 0.f;
}

// ---------------------------------------------------------------------------
// Kpre: pre-swizzle all weight matrices to bf16 WMMA B-fragment layout.
// grid.y = weight id, grid.x = (k-chunk, col-tile) block; one wave per block.
// ---------------------------------------------------------------------------
__global__ __launch_bounds__(32)
void swizzle_weights_kernel(const float* __restrict__ W_r1, const float* __restrict__ W_r2,
                            const float* __restrict__ W1_s, const float* __restrict__ W_sc_s,
                            const float* __restrict__ W1_v, const float* __restrict__ W_sc_v,
                            const float* __restrict__ W2_s, const float* __restrict__ W2_v,
                            __bf16* __restrict__ o_r1, __bf16* __restrict__ o_r2,
                            __bf16* __restrict__ o1s, __bf16* __restrict__ oscs,
                            __bf16* __restrict__ o1v, __bf16* __restrict__ oscv,
                            __bf16* __restrict__ o2s, __bf16* __restrict__ o2v) {
  const float* W; __bf16* out;
  int ldb, Keff, ncols, ntiles, nblocks;
  switch (blockIdx.y) {
    case 0: W = W_r1;   out = o_r1; ldb = 100; Keff = 12;  ncols = 100; ntiles = 7;  nblocks = 7;  break;
    case 1: W = W_r2;   out = o_r2; ldb = 224; Keff = 100; ncols = 224; ntiles = 14; nblocks = 56; break;
    case 2: W = W1_s;   out = o1s;  ldb = 64;  Keff = 64;  ncols = 64;  ntiles = 4;  nblocks = 8;  break;
    case 3: W = W_sc_s; out = oscs; ldb = 96;  Keff = 64;  ncols = 96;  ntiles = 6;  nblocks = 12; break;
    case 4: W = W1_v;   out = o1v;  ldb = 32;  Keff = 32;  ncols = 32;  ntiles = 2;  nblocks = 2;  break;
    case 5: W = W_sc_v; out = oscv; ldb = 32;  Keff = 32;  ncols = 32;  ntiles = 2;  nblocks = 2;  break;
    case 6: W = W2_s;   out = o2s;  ldb = 96;  Keff = 96;  ncols = 96;  ntiles = 6;  nblocks = 18; break;
    default:W = W2_v;   out = o2v;  ldb = 32;  Keff = 128; ncols = 32;  ntiles = 2;  nblocks = 8;  break;
  }
  int bi = blockIdx.x;
  if (bi >= nblocks) return;
  int kk = bi / ntiles, t = bi - kk * ntiles;
  v16bf b = load_b_frag(W, ldb, kk * 32, Keff, t * 16, ncols);
  *(v16bf*)(out + ((size_t)bi * 32 + (threadIdx.x & 31)) * 16) = b;
}

// ---------------------------------------------------------------------------
// K1: node in-projections.  x row = [s(64) | v(32x3 interleaved)]
// jobs (grid.y): 0-3 s1 (K=64), 4-9 sc_s (K=64), 10-15 v1 (c,tile; K=32),
//                16-21 sc_v.
// ---------------------------------------------------------------------------
__global__ __launch_bounds__(32)
void node_in_kernel(const float* __restrict__ x, const float* __restrict__ z,
                    const __bf16* __restrict__ swW1_s, const __bf16* __restrict__ swW_sc_s,
                    const __bf16* __restrict__ swW1_v, const __bf16* __restrict__ swW_sc_v,
                    float* __restrict__ s1, float* __restrict__ v1,
                    float* __restrict__ sc_s, float* __restrict__ sc_v, int N) {
  const int row0 = blockIdx.x * 16;
  const int job  = blockIdx.y;
  const float inv_s64 = 0.125f;        // 1/sqrt(64)
  const float inv_s32 = 0.17677670f;   // 1/sqrt(32)

  if (job < 4) {                                  // s1 = (s*z) @ W1_s
    int col0 = job * 16;
    v8f acc = {};
#pragma unroll
    for (int kk = 0; kk < 2; ++kk) {
      v16bf a = load_a_frag(x, 160, 1, row0, N, kk * 32, 64, z);
      v16bf b = load_b_swz(swW1_s, kk * 4 + job);
      acc = wmma_bf16(a, b, acc);
    }
    store_c(s1, 64, 1, row0, N, col0, acc, inv_s64, nullptr, 0.f);
  } else if (job < 10) {                          // sc_s = (s*z) @ W_sc_s
    int jt = job - 4, col0 = jt * 16;
    v8f acc = {};
#pragma unroll
    for (int kk = 0; kk < 2; ++kk) {
      v16bf a = load_a_frag(x, 160, 1, row0, N, kk * 32, 64, z);
      v16bf b = load_b_swz(swW_sc_s, kk * 6 + jt);
      acc = wmma_bf16(a, b, acc);
    }
    store_c(sc_s, 96, 1, row0, N, col0, acc, inv_s64, nullptr, 0.f);
  } else {                                        // v1 / sc_v per component c
    int jj = (job < 16) ? (job - 10) : (job - 16);
    int c = jj >> 1, tile = jj & 1, col0 = tile * 16;
    const __bf16* W = (job < 16) ? swW1_v : swW_sc_v;
    float* dst      = (job < 16) ? v1     : sc_v;
    v16bf a = load_a_frag(x + 64 + c, 160, 3, row0, N, 0, 32, z);
    v16bf b = load_b_swz(W, tile);
    v8f acc = {};
    acc = wmma_bf16(a, b, acc);
    store_c(dst + c, 96, 3, row0, N, col0, acc, inv_s32, nullptr, 0.f);
  }
}

// ---------------------------------------------------------------------------
// K2: fused edge pipeline. One wave handles 32 edges (2 M-tiles) so each
// B fragment load feeds 2 WMMAs.
//   h = silu(es @ W_r1 / sqrt(12))   (7 B-frags, 14 WMMAs, K=12 pad 32)
//   w = h @ W_r2 / sqrt(100)         (56 B-frags, 112 WMMAs, K=100 pad 128)
//   messages + float atomics into node_s / node_v.
// Never materializes w (358 MB) or mid tensors (768 MB) in HBM.
// ---------------------------------------------------------------------------
#define EPB 32
__global__ __launch_bounds__(32)
void edge_kernel(const float* __restrict__ edge_attr,
                 const float* __restrict__ edge_scalars,
                 const int* __restrict__ edge_src, const int* __restrict__ edge_dst,
                 const __bf16* __restrict__ swW_r1, const __bf16* __restrict__ swW_r2,
                 const float* __restrict__ s1, const float* __restrict__ v1,
                 float* __restrict__ node_s, float* __restrict__ node_v, int E) {
  __shared__ __bf16 h_tile[2][16 * 128];   // [mtile][edge][k], cols 100..127 zero
  __shared__ float  w_tile[2][16 * 224];   // [mtile][edge][col]

  const int e0 = blockIdx.x * EPB;
  const int l = threadIdx.x;
  const int m = l & 15, half = l >> 4;

  __builtin_prefetch(swW_r1, 0, 1);        // global_prefetch_b8 (L2-resident)
  __builtin_prefetch(swW_r2, 0, 1);

  // --- GEMM1 A fragments (K=12: upper 8 elements statically zero) ---
  v16bf a1[2];
#pragma unroll
  for (int mt = 0; mt < 2; ++mt) {
    int row = e0 + mt * 16 + m;
    bool rv = row < E;
#pragma unroll
    for (int i = 0; i < 8; ++i) {
      int k = 8 * half + i;
      a1[mt][i]     = (rv && k < 12) ? (__bf16)edge_scalars[(size_t)row * 12 + k]
                                     : (__bf16)0.f;
      a1[mt][8 + i] = (__bf16)0.f;  // k = 16+8h+i >= 16 > 12
    }
  }
  const float inv_s12 = 0.28867513f;       // 1/sqrt(12)
#pragma unroll
  for (int t = 0; t < 7; ++t) {
    v16bf b = load_b_swz(swW_r1, t);
#pragma unroll
    for (int mt = 0; mt < 2; ++mt) {
      v8f acc = {};
      acc = wmma_bf16(a1[mt], b, acc);
#pragma unroll
      for (int r = 0; r < 8; ++r) {
        int row = r + 8 * half, col = t * 16 + m;
        h_tile[mt][row * 128 + col] = (__bf16)silu_f(acc[r] * inv_s12);
      }
    }
  }
#pragma unroll
  for (int mt = 0; mt < 2; ++mt)            // zero pad cols 112..127
    for (int idx = l; idx < 16 * 16; idx += 32)
      h_tile[mt][(idx >> 4) * 128 + 112 + (idx & 15)] = (__bf16)0.f;
  asm volatile("s_wait_dscnt 0x0" ::: "memory");

  // --- GEMM2 A fragments of h from LDS (contiguous 8-element runs -> b128) ---
  v16bf a2[2][4];
#pragma unroll
  for (int mt = 0; mt < 2; ++mt) {
    const __bf16* hp = &h_tile[mt][m * 128];
#pragma unroll
    for (int kk = 0; kk < 4; ++kk) {
#pragma unroll
      for (int i = 0; i < 8; ++i) {
        a2[mt][kk][i]     = hp[kk * 32 + 8 * half + i];
        a2[mt][kk][8 + i] = hp[kk * 32 + 16 + 8 * half + i];
      }
    }
  }
#pragma unroll
  for (int t = 0; t < 14; ++t) {
    v8f acc0 = {}, acc1 = {};
#pragma unroll
    for (int kk = 0; kk < 4; ++kk) {
      v16bf b = load_b_swz(swW_r2, kk * 14 + t);
      acc0 = wmma_bf16(a2[0][kk], b, acc0);
      acc1 = wmma_bf16(a2[1][kk], b, acc1);
    }
#pragma unroll
    for (int r = 0; r < 8; ++r) {
      int row = r + 8 * half, col = t * 16 + m;
      w_tile[0][row * 224 + col] = acc0[r] * 0.1f;   // 1/sqrt(100)
      w_tile[1][row * 224 + col] = acc1[r] * 0.1f;
    }
  }
  asm volatile("s_wait_dscnt 0x0" ::: "memory");

  // --- messages + scatter (w layout: w1[0:64] w2[64:128] w3[128:160]
  //     w4[160:192] w5[192:224]) ---
  const float inv_sqrt3 = 0.57735027f, inv_sqrt2 = 0.70710678f;
  for (int i = 0; i < EPB; ++i) {
    int e = e0 + i;
    if (e >= E) break;
    int dst = edge_dst[e], src = edge_src[e];
    float se  = edge_attr[(size_t)e * 4 + 0];
    float ve0 = edge_attr[(size_t)e * 4 + 1];
    float ve1 = edge_attr[(size_t)e * 4 + 2];
    float ve2 = edge_attr[(size_t)e * 4 + 3];
    const float* wr = &w_tile[i >> 4][(i & 15) * 224];
    float* nsp = &node_s[(size_t)dst * 96];
    float* nvp = &node_v[(size_t)dst * 384];

    // vector channels: u = lane (0..31)
    {
      int u = l;
      float vv0 = v1[(size_t)src * 96 + u * 3 + 0];
      float vv1 = v1[(size_t)src * 96 + u * 3 + 1];
      float vv2 = v1[(size_t)src * 96 + u * 3 + 2];
      float w3v = wr[128 + u], w4v = wr[160 + u], w5v = wr[192 + u];
      float dot = vv0 * ve0 + vv1 * ve1 + vv2 * ve2;
      atomicAdd(&nsp[64 + u], w4v * dot * inv_sqrt3);              // m0b
      float t3 = w3v * se;
      atomicAdd(&nvp[(64 + u) * 3 + 0], t3 * vv0);                 // m1b
      atomicAdd(&nvp[(64 + u) * 3 + 1], t3 * vv1);
      atomicAdd(&nvp[(64 + u) * 3 + 2], t3 * vv2);
      float cx0 = vv1 * ve2 - vv2 * ve1;                           // m1c
      float cx1 = vv2 * ve0 - vv0 * ve2;
      float cx2 = vv0 * ve1 - vv1 * ve0;
      float t5 = w5v * inv_sqrt2;
      atomicAdd(&nvp[(96 + u) * 3 + 0], t5 * cx0);
      atomicAdd(&nvp[(96 + u) * 3 + 1], t5 * cx1);
      atomicAdd(&nvp[(96 + u) * 3 + 2], t5 * cx2);
    }
    // scalar channels: col = lane, lane+32
#pragma unroll
    for (int p = 0; p < 2; ++p) {
      int col = l + p * 32;
      float ssv = s1[(size_t)src * 64 + col];
      atomicAdd(&nsp[col], wr[col] * ssv * se);                    // m0a
      float t2 = wr[64 + col] * ssv;                               // m1a
      atomicAdd(&nvp[col * 3 + 0], t2 * ve0);
      atomicAdd(&nvp[col * 3 + 1], t2 * ve1);
      atomicAdd(&nvp[col * 3 + 2], t2 * ve2);
    }
  }
}

// ---------------------------------------------------------------------------
// K3: node out-projections + skip blend.
// jobs: 0-5 y_s col tiles (K=96); 6-11 y_v (c,tile; K=128).
// 1/sqrt(16) segment-sum scale folded into the WMMA output scale.
// ---------------------------------------------------------------------------
__global__ __launch_bounds__(32)
void node_out_kernel(const float* __restrict__ node_s, const float* __restrict__ node_v,
                     const float* __restrict__ sc_s, const float* __restrict__ sc_v,
                     const __bf16* __restrict__ swW2_s, const __bf16* __restrict__ swW2_v,
                     const float* __restrict__ z,
                     float* __restrict__ y_s, float* __restrict__ y_v, int N) {
  const int row0 = blockIdx.x * 16;
  const int job  = blockIdx.y;
  const float c_s = 0.38268343f;   // sin(pi/8)
  const float c_x = 0.92387953f;   // cos(pi/8)

  if (job < 6) {
    int col0 = job * 16;
    v8f acc = {};
#pragma unroll
    for (int kk = 0; kk < 3; ++kk) {
      v16bf a = load_a_frag(node_s, 96, 1, row0, N, kk * 32, 96, z);
      v16bf b = load_b_swz(swW2_s, kk * 6 + job);
      acc = wmma_bf16(a, b, acc);
    }
    const float scale = c_x * 0.25f * 0.10206207f;   // c_x/(sqrt(16)*sqrt(96))
    store_c(y_s, 96, 1, row0, N, col0, acc, scale, sc_s, c_s);
  } else {
    int jj = job - 6;
    int c = jj >> 1, tile = jj & 1, col0 = tile * 16;
    v8f acc = {};
#pragma unroll
    for (int kk = 0; kk < 4; ++kk) {
      v16bf a = load_a_frag(node_v + c, 384, 3, row0, N, kk * 32, 128, z);
      v16bf b = load_b_swz(swW2_v, kk * 2 + tile);
      acc = wmma_bf16(a, b, acc);
    }
    const float scale = c_x * 0.25f * 0.08838835f;   // c_x/(sqrt(16)*sqrt(128))
    store_c(y_v + c, 96, 3, row0, N, col0, acc, scale, sc_v + c, c_s);
  }
}

// ---------------------------------------------------------------------------
// K4: silu / sigmoid gating -> out [N,160]
// ---------------------------------------------------------------------------
__global__ void finalize_kernel(const float* __restrict__ y_s,
                                const float* __restrict__ y_v,
                                float* __restrict__ out, int N) {
  int idx = blockIdx.x * blockDim.x + threadIdx.x;
  int total = N * 160;
  if (idx >= total) return;
  int n = idx / 160, i = idx - n * 160;
  if (i < 64) {
    out[idx] = silu_f(y_s[(size_t)n * 96 + i]);
  } else {
    int j = i - 64, u = j / 3;
    float g = 1.f / (1.f + __expf(-y_s[(size_t)n * 96 + 64 + u]));
    out[idx] = y_v[(size_t)n * 96 + j] * g;
  }
}

// ---------------------------------------------------------------------------
extern "C" void kernel_launch(void* const* d_in, const int* in_sizes, int n_in,
                              void* d_out, int out_size, void* d_ws, size_t ws_size,
                              hipStream_t stream) {
  const float* x            = (const float*)d_in[0];
  const float* z            = (const float*)d_in[1];
  const int*   edge_src     = (const int*)d_in[2];
  const int*   edge_dst     = (const int*)d_in[3];
  const float* edge_attr    = (const float*)d_in[4];
  const float* edge_scalars = (const float*)d_in[5];
  const float* W_sc_s       = (const float*)d_in[6];
  const float* W_sc_v       = (const float*)d_in[7];
  const float* W1_s         = (const float*)d_in[8];
  const float* W1_v         = (const float*)d_in[9];
  const float* W_r1         = (const float*)d_in[10];
  const float* W_r2         = (const float*)d_in[11];
  const float* W2_s         = (const float*)d_in[12];
  const float* W2_v         = (const float*)d_in[13];

  const int N = in_sizes[1];   // z: [N,1]
  const int E = in_sizes[2];   // edge_src: [E]

  float* ws     = (float*)d_ws;
  float* s1     = ws;                          // [N,64]
  float* v1     = s1     + (size_t)N * 64;     // [N,32,3]
  float* sc_s   = v1     + (size_t)N * 96;     // [N,96]
  float* sc_v   = sc_s   + (size_t)N * 96;     // [N,32,3]
  float* node_s = sc_v   + (size_t)N * 96;     // [N,96]  accumulator
  float* node_v = node_s + (size_t)N * 96;     // [N,128,3] accumulator
  float* y_s    = node_v + (size_t)N * 384;    // [N,96]
  float* y_v    = y_s    + (size_t)N * 96;     // [N,32,3]

  // pre-swizzled bf16 weight fragments (512 bf16 per 32x16 block)
  __bf16* wb       = (__bf16*)(y_v + (size_t)N * 96);
  __bf16* swW_r1   = wb;                       // 7  blocks
  __bf16* swW_r2   = swW_r1   + 7  * 512;      // 56 blocks
  __bf16* swW1_s   = swW_r2   + 56 * 512;      // 8
  __bf16* swW_sc_s = swW1_s   + 8  * 512;      // 12
  __bf16* swW1_v   = swW_sc_s + 12 * 512;      // 2
  __bf16* swW_sc_v = swW1_v   + 2  * 512;      // 2
  __bf16* swW2_s   = swW_sc_v + 2  * 512;      // 18
  __bf16* swW2_v   = swW2_s   + 18 * 512;      // 8

  const int ntiles = (N + 15) / 16;

  swizzle_weights_kernel<<<dim3(56, 8), 32, 0, stream>>>(
      W_r1, W_r2, W1_s, W_sc_s, W1_v, W_sc_v, W2_s, W2_v,
      swW_r1, swW_r2, swW1_s, swW_sc_s, swW1_v, swW_sc_v, swW2_s, swW2_v);
  zero_kernel<<<(N * 480 + 255) / 256, 256, 0, stream>>>(node_s, N * 480);
  node_in_kernel<<<dim3(ntiles, 22), 32, 0, stream>>>(
      x, z, swW1_s, swW_sc_s, swW1_v, swW_sc_v, s1, v1, sc_s, sc_v, N);
  edge_kernel<<<(E + EPB - 1) / EPB, 32, 0, stream>>>(
      edge_attr, edge_scalars, edge_src, edge_dst, swW_r1, swW_r2,
      s1, v1, node_s, node_v, E);
  node_out_kernel<<<dim3(ntiles, 12), 32, 0, stream>>>(
      node_s, node_v, sc_s, sc_v, swW2_s, swW2_v, z, y_s, y_v, N);
  finalize_kernel<<<((N * 160) + 255) / 256, 256, 0, stream>>>(
      y_s, y_v, (float*)d_out, N);
}
